// SUGModule_27891517620958
// MI455X (gfx1250) — compile-verified, matrix operand order checked by reading, not run on hardware
//
#include <hip/hip_runtime.h>
#include <math.h>

// ---------------------------------------------------------------------------
// SUG GNN forward for MI455X (gfx1250, wave32, WMMA).
// Heavy edge/node MLP GEMMs run on v_wmma_f32_16x16x32_bf16 with fp32
// accumulation; edge messages are fused (never materialized) and scattered
// with global_atomic_add_f32. Edge kernel register-blocks 64 edges per
// workgroup so each weight fragment loaded from L2 feeds 4 WMMAs.
// ---------------------------------------------------------------------------

typedef __attribute__((ext_vector_type(16))) __bf16 v16bf;
typedef __attribute__((ext_vector_type(8)))  __bf16 v8bf;
typedef __attribute__((ext_vector_type(8)))  float  v8f;
typedef unsigned short u16; // raw bf16 bits in buffers (host-safe)

#define N_NODES 10000
#define NPG     1250
#define NGRAPH  8
#define NEDGE   160000
#define DIN     28
#define HDIM    128
#define NBINS   16
#define NLAYERS 6
#define NLYS    1250
#define RMAXF   10.0f

// d_out layout (float elements), reference tuple (sug, ugs, lys_idx, h, lys_summary)
#define OUT_SUG  0
#define OUT_UGS  1024
#define OUT_LIDX 2274
#define OUT_H    3524
#define OUT_LSUM 1283524

// ---------------- device helpers ----------------

__device__ __forceinline__ u16 f2bfu(float f) {
  union { float f; unsigned u; } a; a.f = f;
  return (u16)((a.u + 0x7fffu + ((a.u >> 16) & 1u)) >> 16);
}
__device__ __forceinline__ float siluf(float x) { return x / (1.0f + __expf(-x)); }

// Compose a 16x32 bf16 WMMA fragment from two contiguous 16-byte chunks.
// Element order: e<8 -> VGPR e/2 (k = 8h + e), e>=8 -> VGPR 4+(e-8)/2 (k = 16+8h+(e-8)).
__device__ __forceinline__ v16bf load_frag16(const u16* p0, const u16* p1) {
  v8bf x = *(const v8bf*)p0;
  v8bf y = *(const v8bf*)p1;
  v16bf r;
#pragma unroll
  for (int i = 0; i < 8; ++i) { r[i] = x[i]; r[8 + i] = y[i]; }
  return r;
}
__device__ __forceinline__ v8f wmma_bf16(v16bf a, v16bf b, v8f c) {
  return __builtin_amdgcn_wmma_f32_16x16x32_bf16(false, a, false, b, (short)0, c,
                                                 false, false);
}
__device__ __forceinline__ float wave_sum(float v) {
#pragma unroll
  for (int off = 16; off > 0; off >>= 1) v += __shfl_xor(v, off, 32);
  return v;
}

// ---------------- weight packing: fp32 [K][N] -> bf16 WMMA B-fragments ------
// Layout: frag[(kt*NT + nt)*32 + lane][e], 16 bf16 per lane (contiguous 32B).
// lane = n + 16*h ; k_in_tile = e + 8*h + (e>=8 ? 8 : 0).

__global__ void pack_w_kernel(const float* __restrict__ W, u16* __restrict__ out,
                              int K, int Nout, int Kpad) {
  int idx = blockIdx.x * 256 + threadIdx.x;
  int NT = Nout >> 4;
  int total = (Kpad >> 5) * NT * 512;
  if (idx >= total) return;
  int e = idx & 15;
  int lane = (idx >> 4) & 31;
  int tile = idx >> 9;
  int nt = tile % NT, kt = tile / NT;
  int n = lane & 15, h = lane >> 4;
  int k = kt * 32 + e + 8 * h + ((e >= 8) ? 8 : 0);
  float v = (k < K) ? W[(size_t)k * Nout + nt * 16 + n] : 0.0f;
  out[idx] = f2bfu(v);
}

// ---------------- RBF expansion (bf16) ----------------

__global__ void rbf_kernel(const float* __restrict__ elen, u16* __restrict__ rbf) {
  int idx = blockIdx.x * 256 + threadIdx.x;
  if (idx >= NEDGE * NBINS) return;
  int e = idx >> 4, k = idx & 15;
  float step = RMAXF / (NBINS - 1);
  float c = k * step;
  float w = step * 0.5f;
  float t = elen[e] - c;
  rbf[idx] = f2bfu(__expf(-t * t / (2.0f * w * w)));
}

// ---------------- input embedding + LN ----------------

__global__ __launch_bounds__(128) void embed_kernel(
    const float* __restrict__ x, const float* __restrict__ Wi,
    const float* __restrict__ bi, const float* __restrict__ g,
    const float* __restrict__ b, float* __restrict__ h, u16* __restrict__ hbf) {
  int n = blockIdx.x, f = threadIdx.x;
  float acc = bi[f];
#pragma unroll
  for (int i = 0; i < DIN; ++i) acc += x[n * DIN + i] * Wi[i * HDIM + f];
  float v = siluf(acc);
  __shared__ float red[HDIM];
  red[f] = v;
  __syncthreads();
  for (int s = 64; s > 0; s >>= 1) { if (f < s) red[f] += red[f + s]; __syncthreads(); }
  float mu = red[0] / HDIM;
  __syncthreads();
  float dv = v - mu;
  red[f] = dv * dv;
  __syncthreads();
  for (int s = 64; s > 0; s >>= 1) { if (f < s) red[f] += red[f + s]; __syncthreads(); }
  float rstd = rsqrtf(red[0] / HDIM + 1e-5f);
  float o = dv * rstd * g[f] + b[f];
  h[(size_t)n * HDIM + f] = o;
  hbf[(size_t)n * HDIM + f] = f2bfu(o);
}

// ---------------- degree counts ----------------

__global__ void degree_kernel(const int* __restrict__ ei, float* __restrict__ counts) {
  int e = blockIdx.x * 256 + threadIdx.x;
  if (e >= NEDGE) return;
  atomicAdd(&counts[ei[NEDGE + e]], 1.0f);
}

// ---------------- edge message MLP (fused, WMMA, 64 edges / block) ----------
// A = [h[src](128) | h[dst](128) | rbf(16) | zero-pad(16)]  : 64 edges x 288
// GEMM1 (K=288 -> 256) + SiLU ; GEMM2 (K=256 -> 128) ; atomic scatter to agg.
// B fragments are held in registers and reused across 4 M-subtiles so L2
// weight traffic is amortized 4x (~64 FLOP per L2 byte).

__global__ __launch_bounds__(256) void edge_kernel(
    const int* __restrict__ ei, const u16* __restrict__ hbf,
    const u16* __restrict__ rbf, const u16* __restrict__ W1p,
    const float* __restrict__ b1, const u16* __restrict__ W2p,
    const float* __restrict__ b2, float* __restrict__ agg) {
  __shared__ __align__(16) u16 ash[64][288];
  __shared__ __align__(16) u16 smid[64][256];
  __shared__ int ssrc[64], sdst[64];
  int t = threadIdx.x;
  int e0 = blockIdx.x * 64;
  if (t < 64) { ssrc[t] = ei[e0 + t]; sdst[t] = ei[NEDGE + e0 + t]; }
  __syncthreads();
  // stage A into LDS: 64 rows x 36 chunks of 8 halfs (16B)
  for (int c = t; c < 2304; c += 256) {
    int row = c / 36, ch = c % 36;
    uint4 v;
    if (ch < 16)       v = *(const uint4*)(hbf + (size_t)ssrc[row] * HDIM + ch * 8);
    else if (ch < 32)  v = *(const uint4*)(hbf + (size_t)sdst[row] * HDIM + (ch - 16) * 8);
    else if (ch < 34)  v = *(const uint4*)(rbf + (size_t)(e0 + row) * NBINS + (ch - 32) * 8);
    else               v = make_uint4(0, 0, 0, 0);
    *(uint4*)&ash[row][ch * 8] = v;
  }
  __syncthreads();

  int lane = t & 31, w = t >> 5;
  int m = lane & 15, hh = lane >> 4, n = lane & 15;
  int nt0 = 2 * w, nt1 = 2 * w + 1;
  __builtin_prefetch((const void*)(W2p + (size_t)(w * 32 + lane) * 16), 0, 1);

  v8f acc0[4], acc1[4];
#pragma unroll
  for (int s = 0; s < 4; ++s) {
    acc0[s] = (v8f){0, 0, 0, 0, 0, 0, 0, 0};
    acc1[s] = (v8f){0, 0, 0, 0, 0, 0, 0, 0};
  }
#pragma unroll
  for (int kt = 0; kt < 9; ++kt) {
    const u16* bp0 = W1p + ((size_t)(kt * 16 + nt0) * 32 + lane) * 16;
    const u16* bp1 = W1p + ((size_t)(kt * 16 + nt1) * 32 + lane) * 16;
    v16bf b0v = load_frag16(bp0, bp0 + 8);
    v16bf b1v = load_frag16(bp1, bp1 + 8);
#pragma unroll
    for (int s = 0; s < 4; ++s) {
      const u16* ar = &ash[s * 16 + m][kt * 32 + 8 * hh];
      v16bf a = load_frag16(ar, ar + 16);
      acc0[s] = wmma_bf16(a, b0v, acc0[s]);
      acc1[s] = wmma_bf16(a, b1v, acc1[s]);
    }
  }
  float bia0 = b1[nt0 * 16 + n], bia1 = b1[nt1 * 16 + n];
#pragma unroll
  for (int s = 0; s < 4; ++s) {
#pragma unroll
    for (int r = 0; r < 8; ++r) {
      int mm = s * 16 + r + 8 * hh;
      smid[mm][nt0 * 16 + n] = f2bfu(siluf(acc0[s][r] + bia0));
      smid[mm][nt1 * 16 + n] = f2bfu(siluf(acc1[s][r] + bia1));
    }
  }
  __syncthreads();

  v8f acc2[4];
#pragma unroll
  for (int s = 0; s < 4; ++s) acc2[s] = (v8f){0, 0, 0, 0, 0, 0, 0, 0};
#pragma unroll
  for (int kt = 0; kt < 8; ++kt) {
    const u16* bp = W2p + ((size_t)(kt * 8 + w) * 32 + lane) * 16;
    v16bf bv = load_frag16(bp, bp + 8);
#pragma unroll
    for (int s = 0; s < 4; ++s) {
      const u16* ar = &smid[s * 16 + m][kt * 32 + 8 * hh];
      acc2[s] = wmma_bf16(load_frag16(ar, ar + 16), bv, acc2[s]);
    }
  }
  int col = w * 16 + n;
  float bias = b2[col];
#pragma unroll
  for (int s = 0; s < 4; ++s) {
#pragma unroll
    for (int r = 0; r < 8; ++r) {
      int mm = s * 16 + r + 8 * hh;
      atomicAdd(&agg[(size_t)sdst[mm] * HDIM + col], acc2[s][r] + bias);
    }
  }
}

// ---------------- node update MLP + LN + residual (WMMA) ----------------

__global__ __launch_bounds__(256) void node_kernel(
    const u16* __restrict__ hbfin, const float* __restrict__ agg,
    const float* __restrict__ counts, const u16* __restrict__ U1p,
    const float* __restrict__ b1, const u16* __restrict__ U2p,
    const float* __restrict__ b2, const float* __restrict__ lng,
    const float* __restrict__ lnb, float* __restrict__ h, u16* __restrict__ hbfout) {
  __shared__ __align__(16) u16 ash[16][256];
  __shared__ __align__(16) u16 smid[16][128];
  __shared__ float of[16][128];
  __shared__ float scnt[16];
  int t = threadIdx.x;
  int n0 = blockIdx.x * 16;
  if (t < 16) scnt[t] = fmaxf(counts[n0 + t], 1.0f);
  __syncthreads();
  for (int c = t; c < 512; c += 256) {
    int row = c >> 5, ch = c & 31;
    int node = n0 + row;
    if (ch < 16) {
      uint4 v = *(const uint4*)(hbfin + (size_t)node * HDIM + ch * 8);
      *(uint4*)&ash[row][ch * 8] = v;
    } else {
      float inv = 1.0f / scnt[row];
      int base = (ch - 16) * 8;
#pragma unroll
      for (int j = 0; j < 8; ++j)
        ash[row][HDIM + base + j] = f2bfu(agg[(size_t)node * HDIM + base + j] * inv);
    }
  }
  __syncthreads();

  int lane = t & 31, w = t >> 5;
  int m = lane & 15, hh = lane >> 4, n = lane & 15;
  v8f c0 = {0, 0, 0, 0, 0, 0, 0, 0};
#pragma unroll
  for (int kt = 0; kt < 8; ++kt) {
    const u16* ar = &ash[m][kt * 32 + 8 * hh];
    const u16* bp = U1p + ((size_t)(kt * 8 + w) * 32 + lane) * 16;
    c0 = wmma_bf16(load_frag16(ar, ar + 16), load_frag16(bp, bp + 8), c0);
  }
#pragma unroll
  for (int r = 0; r < 8; ++r) {
    int mm = r + 8 * hh;
    smid[mm][w * 16 + n] = f2bfu(siluf(c0[r] + b1[w * 16 + n]));
  }
  __syncthreads();

  v8f acc = {0, 0, 0, 0, 0, 0, 0, 0};
#pragma unroll
  for (int kt = 0; kt < 4; ++kt) {
    const u16* ar = &smid[m][kt * 32 + 8 * hh];
    const u16* bp = U2p + ((size_t)(kt * 8 + w) * 32 + lane) * 16;
    acc = wmma_bf16(load_frag16(ar, ar + 16), load_frag16(bp, bp + 8), acc);
  }
#pragma unroll
  for (int r = 0; r < 8; ++r) {
    int mm = r + 8 * hh;
    of[mm][w * 16 + n] = acc[r] + b2[w * 16 + n];
  }
  __syncthreads();

  // LN + residual: wave w handles rows 2w, 2w+1 (wave32 shuffle reduction)
#pragma unroll
  for (int rr = 0; rr < 2; ++rr) {
    int row = 2 * w + rr;
    float sum = 0.0f, sq = 0.0f;
#pragma unroll
    for (int j = 0; j < 4; ++j) {
      float v = of[row][lane + 32 * j];
      sum += v; sq += v * v;
    }
    sum = wave_sum(sum); sq = wave_sum(sq);
    float mu = sum / HDIM;
    float var = sq / HDIM - mu * mu;
    float rstd = rsqrtf(var + 1e-5f);
    int node = n0 + row;
#pragma unroll
    for (int j = 0; j < 4; ++j) {
      int cc = lane + 32 * j;
      float v = (of[row][cc] - mu) * rstd * lng[cc] + lnb[cc];
      float hn = h[(size_t)node * HDIM + cc] + v;
      h[(size_t)node * HDIM + cc] = hn;
      hbfout[(size_t)node * HDIM + cc] = f2bfu(hn);
    }
  }
}

// ---------------- lysine index extraction (stable nonzero) ----------------

__global__ void lys_count_kernel(const int* __restrict__ mask, int* __restrict__ bc) {
  __shared__ int red[256];
  int t = threadIdx.x, nidx = blockIdx.x * 256 + t;
  red[t] = (nidx < N_NODES && mask[nidx] != 0) ? 1 : 0;
  __syncthreads();
  for (int s = 128; s > 0; s >>= 1) { if (t < s) red[t] += red[t + s]; __syncthreads(); }
  if (t == 0) bc[blockIdx.x] = red[0];
}
__global__ void lys_scan_kernel(const int* __restrict__ bc, int* __restrict__ bo, int nb) {
  if (threadIdx.x == 0 && blockIdx.x == 0) {
    int a = 0;
    for (int i = 0; i < nb; ++i) { bo[i] = a; a += bc[i]; }
  }
}
__global__ void lys_write_kernel(const int* __restrict__ mask, const int* __restrict__ bo,
                                 int* __restrict__ lysidx, int* __restrict__ outIdx) {
  __shared__ int wcnt[8], woff[8];
  int t = threadIdx.x, nidx = blockIdx.x * 256 + t;
  int lane = t & 31, w = t >> 5;
  int m = (nidx < N_NODES && mask[nidx] != 0) ? 1 : 0;
  unsigned long long bal = __ballot(m);
  if (lane == 0) wcnt[w] = __popcll(bal);
  __syncthreads();
  if (t == 0) { int a = 0; for (int i = 0; i < 8; ++i) { woff[i] = a; a += wcnt[i]; } }
  __syncthreads();
  if (m) {
    int pre = __popcll(bal & ((1ull << lane) - 1ull));
    int idx = bo[blockIdx.x] + woff[w] + pre;
    if (idx < NLYS) { lysidx[idx] = nidx; outIdx[idx] = nidx; }
  }
}

// ---------------- per-lysine scoring + token MLP ----------------

__global__ __launch_bounds__(128) void lys_mlp_kernel(
    const int* __restrict__ lysidx, const float* __restrict__ h,
    const float* __restrict__ sc1w, const float* __restrict__ sc1b,
    const float* __restrict__ sc2w, const float* __restrict__ sc2b,
    const float* __restrict__ sc3w, const float* __restrict__ sc3b,
    const float* __restrict__ ls1w, const float* __restrict__ ls1b,
    const float* __restrict__ ls2w, const float* __restrict__ ls2b,
    float* __restrict__ ugs, float* __restrict__ lt) {
  __shared__ float lf[128], t1[128], t2[64], t3[128];
  __shared__ float su;
  int i = blockIdx.x, f = threadIdx.x;
  int node = lysidx[i];
  lf[f] = h[(size_t)node * HDIM + f];
  __syncthreads();
  float a1 = sc1b[f];
  for (int k = 0; k < 128; ++k) a1 += lf[k] * sc1w[k * 128 + f];
  t1[f] = siluf(a1);
  __syncthreads();
  if (f < 64) {
    float a2 = sc2b[f];
    for (int k = 0; k < 128; ++k) a2 += t1[k] * sc2w[k * 64 + f];
    t2[f] = siluf(a2);
  }
  __syncthreads();
  if (f == 0) {
    float a3 = sc3b[0];
    for (int k = 0; k < 64; ++k) a3 += t2[k] * sc3w[k];
    float u = 1.0f / (1.0f + __expf(-a3));
    ugs[i] = u; su = u;
  }
  __syncthreads();
  float a4 = ls1b[f];
  for (int k = 0; k < 128; ++k) a4 += lf[k] * ls1w[k * 128 + f];
  a4 += su * ls1w[128 * 128 + f];
  t3[f] = siluf(a4);
  __syncthreads();
  float a5 = ls2b[f];
  for (int k = 0; k < 128; ++k) a5 += t3[k] * ls2w[k * 128 + f];
  lt[(size_t)i * 128 + f] = a5;
}

// ---------------- graph pooling ----------------

__global__ __launch_bounds__(128) void pool_kernel(const float* __restrict__ h,
                                                   const int* __restrict__ batch,
                                                   float* __restrict__ gp_in) {
  int g = blockIdx.x, f = threadIdx.x;
  float s = 0.0f, mx = -3.4e38f, cnt = 0.0f;
  for (int n = 0; n < N_NODES; ++n) {
    if (batch[n] == g) {
      float v = h[(size_t)n * HDIM + f];
      s += v; mx = fmaxf(mx, v); cnt += 1.0f;
    }
  }
  float sn = fmaxf(sqrtf(cnt), 1.0f);
  gp_in[g * 256 + f] = (s / cnt) / sn;
  gp_in[g * 256 + 128 + f] = mx;
}

__global__ __launch_bounds__(1024) void gp_kernel(
    const float* __restrict__ gp_in, const float* __restrict__ gw,
    const float* __restrict__ gb, const float* __restrict__ lng,
    const float* __restrict__ lnb, float* __restrict__ sug) {
  __shared__ float sh[8][128];
  int t = threadIdx.x;
  int g = t >> 7, f = t & 127;
  float acc = gb[f];
  for (int k = 0; k < 256; ++k) acc += gp_in[g * 256 + k] * gw[k * 128 + f];
  sh[g][f] = siluf(acc);
  __syncthreads();
  int w = t >> 5, lane = t & 31;
  if (w < 8) {
    float sum = 0.0f, sq = 0.0f;
#pragma unroll
    for (int j = 0; j < 4; ++j) { float v = sh[w][lane + 32 * j]; sum += v; sq += v * v; }
    sum = wave_sum(sum); sq = wave_sum(sq);
    float mu = sum / 128.0f;
    float rstd = rsqrtf(sq / 128.0f - mu * mu + 1e-5f);
#pragma unroll
    for (int j = 0; j < 4; ++j) {
      int cc = lane + 32 * j;
      sug[w * 128 + cc] = (sh[w][cc] - mu) * rstd * lng[cc] + lnb[cc];
    }
  }
}

// ---------------- softmax-weighted lysine summary ----------------

__global__ __launch_bounds__(128) void summary_kernel(
    const float* __restrict__ ugs, const float* __restrict__ lt,
    const int* __restrict__ lysidx, const int* __restrict__ batch,
    float* __restrict__ lsum) {
  __shared__ float red[128];
  __shared__ float smx, sden;
  int g = blockIdx.x, t = threadIdx.x;
  float mx = -3.4e38f;
  for (int i = t; i < NLYS; i += 128)
    if (batch[lysidx[i]] == g) mx = fmaxf(mx, ugs[i]);
  red[t] = mx;
  __syncthreads();
  for (int s = 64; s > 0; s >>= 1) { if (t < s) red[t] = fmaxf(red[t], red[t + s]); __syncthreads(); }
  if (t == 0) smx = red[0];
  __syncthreads();
  float ds = 0.0f;
  for (int i = t; i < NLYS; i += 128)
    if (batch[lysidx[i]] == g) ds += __expf(ugs[i] - smx);
  red[t] = ds;
  __syncthreads();
  for (int s = 64; s > 0; s >>= 1) { if (t < s) red[t] += red[t + s]; __syncthreads(); }
  if (t == 0) sden = red[0];
  __syncthreads();
  float acc = 0.0f;
  for (int i = 0; i < NLYS; ++i)
    if (batch[lysidx[i]] == g)
      acc += __expf(ugs[i] - smx) / sden * lt[(size_t)i * 128 + t];
  lsum[g * 128 + t] = acc;
}

// ---------------- host orchestration ----------------
// Input flattening assumption (setup_inputs insertion order):
// 0:x 1:edge_index 2:edge_vec 3:edge_len 4:lysine_mask 5:batch
// 6,7:inp.w/b 8,9:inp_ln.g/b
// 10+10l .. : per layer m1.w,m1.b,m2.w,m2.b,u1.w,u1.b,u2.w,u2.b,ln.g,ln.b
// 70..75: sc1.w/b sc2.w/b sc3.w/b  76,77:gp.w/b 78,79:gp_ln.g/b
// 80..83: ls1.w/b ls2.w/b

extern "C" void kernel_launch(void* const* d_in, const int* in_sizes, int n_in,
                              void* d_out, int out_size, void* d_ws, size_t ws_size,
                              hipStream_t stream) {
  const float* x     = (const float*)d_in[0];
  const int*   ei    = (const int*)d_in[1];
  const float* elen  = (const float*)d_in[3];
  const int*   mask  = (const int*)d_in[4];
  const int*   batch = (const int*)d_in[5];
  const float* inpw = (const float*)d_in[6];
  const float* inpb = (const float*)d_in[7];
  const float* ilng = (const float*)d_in[8];
  const float* ilnb = (const float*)d_in[9];
  auto LP = [&](int l, int j) { return (const float*)d_in[10 + l * 10 + j]; };
  const float* sc1w = (const float*)d_in[70]; const float* sc1b = (const float*)d_in[71];
  const float* sc2w = (const float*)d_in[72]; const float* sc2b = (const float*)d_in[73];
  const float* sc3w = (const float*)d_in[74]; const float* sc3b = (const float*)d_in[75];
  const float* gpw  = (const float*)d_in[76]; const float* gpb  = (const float*)d_in[77];
  const float* glng = (const float*)d_in[78]; const float* glnb = (const float*)d_in[79];
  const float* ls1w = (const float*)d_in[80]; const float* ls1b = (const float*)d_in[81];
  const float* ls2w = (const float*)d_in[82]; const float* ls2b = (const float*)d_in[83];

  // workspace layout (bytes); total ~15.4 MB
  char* ws = (char*)d_ws;
  u16*   hbf    = (u16*)(ws);                       // N*128*2      = 2,560,000
  u16*   rbfbf  = (u16*)(ws + 2560000);             // E*16*2       = 5,120,000
  float* agg    = (float*)(ws + 7680000);           // N*128*4      = 5,120,000
  float* counts = (float*)(ws + 12800000);          // N*4          = 40,000
  u16*   wpack  = (u16*)(ws + 12840000);            // 6*155648*2   = 1,867,776
  int*   lysidx = (int*)(ws + 14707776);            // 1250*4
  float* lt     = (float*)(ws + 14712832);          // 1250*128*4   = 640,000
  float* gp_in  = (float*)(ws + 15352832);          // 8*256*4
  int*   blkcnt = (int*)(ws + 15361024);
  int*   blkoff = blkcnt + 64;

  float* out  = (float*)d_out;
  float* sug  = out + OUT_SUG;
  float* ugs  = out + OUT_UGS;
  int*   lidx = (int*)(out + OUT_LIDX);
  float* h    = out + OUT_H;
  float* lsum = out + OUT_LSUM;

  hipMemsetAsync(counts, 0, N_NODES * sizeof(float), stream);

  // pack all layer weights to bf16 WMMA fragments (per layer: W1 288x256,
  // W2 256x128, U1 256x128, U2 128x128)
  const size_t W1SZ = 73728, W2SZ = 32768, U1SZ = 32768, U2SZ = 16384;
  for (int l = 0; l < NLAYERS; ++l) {
    u16* w1 = wpack + (size_t)l * (W1SZ + W2SZ + U1SZ + U2SZ);
    u16* w2 = w1 + W1SZ;
    u16* u1 = w2 + W2SZ;
    u16* u2 = u1 + U1SZ;
    pack_w_kernel<<<(int)(W1SZ / 256), 256, 0, stream>>>(LP(l, 0), w1, 272, 256, 288);
    pack_w_kernel<<<(int)(W2SZ / 256), 256, 0, stream>>>(LP(l, 2), w2, 256, 128, 256);
    pack_w_kernel<<<(int)(U1SZ / 256), 256, 0, stream>>>(LP(l, 4), u1, 256, 128, 256);
    pack_w_kernel<<<(int)(U2SZ / 256), 256, 0, stream>>>(LP(l, 6), u2, 128, 128, 128);
  }

  rbf_kernel<<<(NEDGE * NBINS + 255) / 256, 256, 0, stream>>>(elen, rbfbf);
  embed_kernel<<<N_NODES, 128, 0, stream>>>(x, inpw, inpb, ilng, ilnb, h, hbf);
  degree_kernel<<<(NEDGE + 255) / 256, 256, 0, stream>>>(ei, counts);

  for (int l = 0; l < NLAYERS; ++l) {
    u16* w1 = wpack + (size_t)l * (W1SZ + W2SZ + U1SZ + U2SZ);
    u16* w2 = w1 + W1SZ;
    u16* u1 = w2 + W2SZ;
    u16* u2 = u1 + U1SZ;
    hipMemsetAsync(agg, 0, (size_t)N_NODES * HDIM * sizeof(float), stream);
    edge_kernel<<<NEDGE / 64, 256, 0, stream>>>(ei, hbf, rbfbf, w1, LP(l, 1), w2,
                                                LP(l, 3), agg);
    node_kernel<<<N_NODES / 16, 256, 0, stream>>>(hbf, agg, counts, u1, LP(l, 5), u2,
                                                  LP(l, 7), LP(l, 8), LP(l, 9), h, hbf);
  }

  lys_count_kernel<<<40, 256, 0, stream>>>(mask, blkcnt);
  lys_scan_kernel<<<1, 32, 0, stream>>>(blkcnt, blkoff, 40);
  lys_write_kernel<<<40, 256, 0, stream>>>(mask, blkoff, lysidx, lidx);
  lys_mlp_kernel<<<NLYS, 128, 0, stream>>>(lysidx, h, sc1w, sc1b, sc2w, sc2b, sc3w,
                                           sc3b, ls1w, ls1b, ls2w, ls2b, ugs, lt);
  pool_kernel<<<NGRAPH, 128, 0, stream>>>(h, batch, gp_in);
  gp_kernel<<<1, 1024, 0, stream>>>(gp_in, gpw, gpb, glng, glnb, sug);
  summary_kernel<<<NGRAPH, 128, 0, stream>>>(ugs, lt, lysidx, batch, lsum);
}
